// DeformConv_1666447311439
// MI455X (gfx1250) — matrix operand order repaired and checked by reading, not compiled, not run on hardware
//
#include <hip/hip_runtime.h>
#include <hip/hip_bf16.h>

#define BB   4
#define CIN  256
#define COUT 256
#define HH   128
#define WW   128
#define HW   (HH*WW)        // 16384
#define KTOT (CIN*9)        // 2304, multiple of 32
#define KC   32             // K per WMMA chunk (bf16 16x16x32)
#define NT   64             // pixels per block tile
#define SK   40             // sB K-stride in bf16 elems (80B: 16B-aligned, bank-spread)

typedef __attribute__((ext_vector_type(16))) __bf16 v16bf;
typedef __attribute__((ext_vector_type(8)))  __bf16 bf16x8;
typedef __attribute__((ext_vector_type(8)))  float  v8f;

// ---------------------------------------------------------------------------
// Kernel A: pack main-conv weights f32 [O][C][3][3] -> bf16 [O][C*9]
// (flat inner index is already c*9+tap, matching GEMM K order)
// ---------------------------------------------------------------------------
__global__ __launch_bounds__(256) void pack_weights(const float* __restrict__ w,
                                                    __bf16* __restrict__ wbf, int n)
{
    int i = blockIdx.x * blockDim.x + threadIdx.x;
    if (i < n) wbf[i] = (__bf16)w[i];
}

// ---------------------------------------------------------------------------
// Kernel B: offset/mask conv (27ch, 3x3) + bilinear metadata.
// One thread per (b, tap k, pixel p). Emits, per sample point:
//   metaI = {clamped base index y0*W+x0, dx (0/1), dy*W (0/W), pad}
//   metaW = 4 corner weights premultiplied by sigmoid(mask) and validity.
// Channel-independent => reused by all 256 input channels in the GEMM.
// ---------------------------------------------------------------------------
__global__ __launch_bounds__(256) void dcn_offset_meta(
    const float* __restrict__ x, const float* __restrict__ offw,
    const float* __restrict__ offb,
    int4* __restrict__ metaI, float4* __restrict__ metaW)
{
    int gid = blockIdx.x * blockDim.x + threadIdx.x;   // (b*9+k)*HW + p
    if (gid >= BB * 9 * HW) return;
    int p  = gid & (HW - 1);
    int bk = gid >> 14;                                // HW = 2^14
    int k  = bk % 9;
    int b  = bk / 9;
    int yy = p >> 7;                                   // W = 128
    int xx = p & 127;

    float sdy = offb[k], sdx = offb[9 + k], sml = offb[18 + k];
    const float* xb = x + (size_t)b * CIN * HW;
    for (int c = 0; c < CIN; ++c) {
        const float* xc  = xb + c * HW;
        const float* wdy = offw + (( k       * CIN + c) * 9);
        const float* wdx = offw + (((9  + k) * CIN + c) * 9);
        const float* wml = offw + (((18 + k) * CIN + c) * 9);
        #pragma unroll
        for (int ty = 0; ty < 3; ++ty) {
            int sy = yy + ty - 1;
            if (sy < 0 || sy >= HH) continue;
            #pragma unroll
            for (int tx = 0; tx < 3; ++tx) {
                int sx = xx + tx - 1;
                if (sx < 0 || sx >= WW) continue;
                float xv = xc[sy * WW + sx];
                int j = ty * 3 + tx;
                sdy = fmaf(xv, wdy[j], sdy);
                sdx = fmaf(xv, wdx[j], sdx);
                sml = fmaf(xv, wml[j], sml);
            }
        }
    }
    float m  = 1.0f / (1.0f + expf(-sml));             // sigmoid(mask)
    float py = (float)yy + (float)(k / 3 - 1) + sdy;
    float px = (float)xx + (float)(k % 3 - 1) + sdx;
    float fy = floorf(py), fx = floorf(px);
    float wy = py - fy,    wx = px - fx;
    int iy0 = (int)fy, ix0 = (int)fx;
    int iy1 = iy0 + 1, ix1 = ix0 + 1;
    float v00 = (iy0 >= 0 && iy0 < HH && ix0 >= 0 && ix0 < WW) ? 1.f : 0.f;
    float v01 = (iy0 >= 0 && iy0 < HH && ix1 >= 0 && ix1 < WW) ? 1.f : 0.f;
    float v10 = (iy1 >= 0 && iy1 < HH && ix0 >= 0 && ix0 < WW) ? 1.f : 0.f;
    float v11 = (iy1 >= 0 && iy1 < HH && ix1 >= 0 && ix1 < WW) ? 1.f : 0.f;
    int y0c = min(max(iy0, 0), HH - 1), y1c = min(max(iy1, 0), HH - 1);
    int x0c = min(max(ix0, 0), WW - 1), x1c = min(max(ix1, 0), WW - 1);
    metaI[gid] = make_int4(y0c * WW + x0c, x1c - x0c, (y1c - y0c) * WW, 0);
    metaW[gid] = make_float4((1.f - wy) * (1.f - wx) * m * v00,
                             (1.f - wy) *  wx        * m * v01,
                              wy        * (1.f - wx) * m * v10,
                              wy        *  wx        * m * v11);
}

// ---------------------------------------------------------------------------
// Kernel C: fused sample + bf16 WMMA GEMM + bias + BN + ReLU.
// Block: 256 threads (8 waves). Tile: all 256 O-channels x 64 pixels.
// Wave w owns M rows [32w, 32w+32) as two 16x16 WMMA M-tiles x 4 N-tiles.
// ---------------------------------------------------------------------------
__global__ __launch_bounds__(256) void dcn_wmma_gemm(
    const float*  __restrict__ x,
    const __bf16* __restrict__ wbf,
    const int4*   __restrict__ metaI,
    const float4* __restrict__ metaW,
    const float*  __restrict__ bias,
    const float*  __restrict__ gamma,
    const float*  __restrict__ beta,
    const float*  __restrict__ rmean,
    const float*  __restrict__ rvar,
    float*        __restrict__ out)
{
    __shared__ int    s_idx[9 * NT];
    __shared__ int    s_dx [9 * NT];
    __shared__ int    s_dyW[9 * NT];
    __shared__ float4 s_w  [9 * NT];
    __shared__ __attribute__((aligned(16))) __bf16 sB[NT * SK];
    __shared__ float  s_sc[COUT], s_sh[COUT];

    const int tid  = threadIdx.x;
    const int lane = tid & 31;
    const int wave = tid >> 5;
    const int blk  = blockIdx.x;
    const int b    = blk >> 8;                  // HW/NT = 256 tiles per batch
    const int p0   = (blk & 255) * NT;

    // per-channel BN+bias fused scale/shift:  out = relu(gemm*sc + sh)
    {
        float sc = gamma[tid] * rsqrtf(rvar[tid] + 1e-5f);
        s_sc[tid] = sc;
        s_sh[tid] = (bias[tid] - rmean[tid]) * sc + beta[tid];
    }
    // stage sampling metadata for this pixel tile (9 taps x 64 pixels)
    for (int e = tid; e < 9 * NT; e += 256) {
        int k = e >> 6, n = e & 63;
        int g = (b * 9 + k) * HW + p0 + n;
        int4 a = metaI[g];
        s_idx[e] = a.x; s_dx[e] = a.y; s_dyW[e] = a.z;
        s_w[e]   = metaW[g];
    }
    __syncthreads();

    v8f acc[2][4] = {};

    const float*  xb    = x + (size_t)b * CIN * HW;
    const int     aKoff = (lane < 16) ? 0 : 8;   // 16-bit A VGPR layout
    const int     bKoff = (lane < 16) ? 0 : 8;   // 16-bit B VGPR layout
    const int     mrow0 = wave * 32 + (lane & 15);
    const __bf16* aRow0 = wbf + (size_t)(mrow0)      * KTOT;
    const __bf16* aRow1 = wbf + (size_t)(mrow0 + 16) * KTOT;

    for (int k0 = 0; k0 < KTOT; k0 += KC) {
        // Build B chunk [32 K x 64 N] bf16 in LDS via fused bilinear gather.
        #pragma unroll
        for (int i = 0; i < 8; ++i) {
            int e   = i * 256 + tid;
            int kkL = e & 31;
            int n   = e >> 5;
            int kk  = k0 + kkL;
            int c   = kk / 9;
            int tap = kk - c * 9;
            int me  = tap * NT + n;
            const float* base = xb + c * HW + s_idx[me];
            int dx = s_dx[me], dyW = s_dyW[me];
            float4 wv = s_w[me];
            float v = wv.x * base[0]   + wv.y * base[dx]
                    + wv.z * base[dyW] + wv.w * base[dyW + dx];
            sB[n * SK + kkL] = (__bf16)v;
        }
        __syncthreads();

        // A fragments: two b128 global loads per M-tile (L2-resident weights)
        v16bf aF[2];
        {
            bf16x8 lo0 = *(const bf16x8*)(aRow0 + k0 + aKoff);
            bf16x8 hi0 = *(const bf16x8*)(aRow0 + k0 + aKoff + 16);
            bf16x8 lo1 = *(const bf16x8*)(aRow1 + k0 + aKoff);
            bf16x8 hi1 = *(const bf16x8*)(aRow1 + k0 + aKoff + 16);
            #pragma unroll
            for (int i = 0; i < 8; ++i) {
                aF[0][i] = lo0[i]; aF[0][i + 8] = hi0[i];
                aF[1][i] = lo1[i]; aF[1][i + 8] = hi1[i];
            }
        }
        #pragma unroll
        for (int n = 0; n < 4; ++n) {
            const __bf16* bp = sB + (n * 16 + (lane & 15)) * SK + bKoff;
            bf16x8 blo = *(const bf16x8*)(bp);
            bf16x8 bhi = *(const bf16x8*)(bp + 16);
            v16bf bF;
            #pragma unroll
            for (int i = 0; i < 8; ++i) { bF[i] = blo[i]; bF[i + 8] = bhi[i]; }
            acc[0][n] = __builtin_amdgcn_wmma_f32_16x16x32_bf16(
                false, aF[0], false, bF, (short)0, acc[0][n], false, false);
            acc[1][n] = __builtin_amdgcn_wmma_f32_16x16x32_bf16(
                false, aF[1], false, bF, (short)0, acc[1][n], false, false);
        }
        __syncthreads();
    }

    // Epilogue: C layout = VGPR r -> M=r (lanes<16) / M=r+8 (lanes>=16), N=lane&15
    const int nBase = p0 + (lane & 15);
    const int oHalf = (lane & 16) ? 8 : 0;
    #pragma unroll
    for (int m = 0; m < 2; ++m) {
        int oBase = wave * 32 + m * 16 + oHalf;
        #pragma unroll
        for (int n = 0; n < 4; ++n) {
            int p = nBase + n * 16;
            #pragma unroll
            for (int r = 0; r < 8; ++r) {
                int o = oBase + r;
                float v = acc[m][n][r] * s_sc[o] + s_sh[o];
                v = v > 0.f ? v : 0.f;
                out[((size_t)b * COUT + o) * HW + p] = v;
            }
        }
    }
}

// ---------------------------------------------------------------------------
extern "C" void kernel_launch(void* const* d_in, const int* in_sizes, int n_in,
                              void* d_out, int out_size, void* d_ws, size_t ws_size,
                              hipStream_t stream)
{
    const float* x     = (const float*)d_in[0];
    const float* offw  = (const float*)d_in[1];
    const float* offb  = (const float*)d_in[2];
    const float* wgt   = (const float*)d_in[3];
    const float* bias  = (const float*)d_in[4];
    const float* gamma = (const float*)d_in[5];
    const float* beta  = (const float*)d_in[6];
    const float* rmean = (const float*)d_in[7];
    const float* rvar  = (const float*)d_in[8];

    const int nMeta = BB * 9 * HW;                       // 589824 sample points
    int4*   metaI = (int4*)d_ws;                         // 9.4 MB
    float4* metaW = (float4*)((char*)d_ws + (size_t)nMeta * 16);   // 9.4 MB
    __bf16* wbf   = (__bf16*)((char*)d_ws + (size_t)nMeta * 32);   // 1.2 MB

    pack_weights<<<(COUT * KTOT + 255) / 256, 256, 0, stream>>>(wgt, wbf, COUT * KTOT);
    dcn_offset_meta<<<(nMeta + 255) / 256, 256, 0, stream>>>(x, offw, offb, metaI, metaW);
    dcn_wmma_gemm<<<BB * (HW / NT), 256, 0, stream>>>(x, wbf, metaI, metaW,
                                                      bias, gamma, beta, rmean, rvar,
                                                      (float*)d_out);
}
